// DualEncoderEpsNetwork_23089744183313
// MI455X (gfx1250) — compile-verified, matrix operand order checked by reading, not run on hardware
//
#include <hip/hip_runtime.h>
#include <hip/hip_bf16.h>

// ---------------------------------------------------------------------------
// Vector types
// ---------------------------------------------------------------------------
typedef __attribute__((ext_vector_type(16))) __bf16 v16bf;
typedef __attribute__((ext_vector_type(8)))  __bf16 v8bf;
typedef __attribute__((ext_vector_type(8)))  float  v8f;
typedef __attribute__((ext_vector_type(4)))  unsigned v4u;
typedef __attribute__((ext_vector_type(8)))  int    v8i;
typedef __attribute__((ext_vector_type(4)))  int    v4i;

#define HDIM 256
#define FDIM 28
#define LAYERS 6
#define TILE_M 128
#define TILE_N 128
#define KSTR 40      // LDS K-stride per W column (32 + 8 pad) -> conflict-free b128 reads

#ifndef __has_builtin
#define __has_builtin(x) 0
#endif
#if __has_builtin(__builtin_amdgcn_tensor_load_to_lds) && \
    __has_builtin(__builtin_amdgcn_s_wait_tensorcnt)
#define USE_TDM 1
#else
#define USE_TDM 0
#endif

enum { ACT_NONE = 0, ACT_SSP = 1, ACT_RELU = 2 };

__device__ __forceinline__ float sspf(float x) {
    float ax = fabsf(x);
    return fmaxf(x, 0.0f) + log1pf(__expf(-ax)) - 0.6931471805599453f;
}

#if USE_TDM
// ---------------------------------------------------------------------------
// Issue one TDM descriptor: DMA a 32(K) x 128(cols) bf16 tile of W^T from
// global into LDS, with hardware padding reproducing the KSTR=40 layout
// (pad 4 DWORDs after every 16 DWORDs = 32 bf16). Tracked by TENSORcnt.
// ---------------------------------------------------------------------------
__device__ __forceinline__ void tdm_issue_w_tile(unsigned lds_byte_addr,
                                                 unsigned long long gaddr,
                                                 int K, int Ncols)
{
    v4u g0;
    g0[0] = 1u;                                   // count=1: one valid descriptor
    g0[1] = lds_byte_addr;                        // LDS destination (bytes)
    g0[2] = (unsigned)gaddr;                      // global_addr[31:0]
    g0[3] = (unsigned)((gaddr >> 32) & 0x1FFFFFFull) | (2u << 30); // hi + type=2
    v8i g1;
    g1[0] = (1 << 16)      // data_size = 2 bytes (bf16)
          | (1 << 20)      // pad_enable
          | (3 << 22)      // pad_interval: 16 DWORDs (= 32 bf16 = tile_dim0)
          | (3 << 25);     // pad_amount: 4 DWORDs (= 8 bf16)  -> KSTR = 40
    g1[1] = (K & 0xFFFF) << 16;                             // tensor_dim0[15:0]
    g1[2] = (int)(((unsigned)K >> 16) | ((unsigned)(Ncols & 0xFFFF) << 16));
    g1[3] = (32 << 16);                                     // tile_dim0 = 32
    g1[4] = TILE_N;                                         // tile_dim1 = 128
    g1[5] = K;                                              // tensor_dim0_stride
    g1[6] = 0;
    g1[7] = 0;
    v4i gz = {0, 0, 0, 0};
#if __clang_major__ >= 23
    v8i gz8 = {0, 0, 0, 0, 0, 0, 0, 0};
    __builtin_amdgcn_tensor_load_to_lds(g0, g1, gz, gz, gz8, 0);
#else
    __builtin_amdgcn_tensor_load_to_lds(g0, g1, gz, gz, 0);
#endif
}
#endif

// ---------------------------------------------------------------------------
// Weight convert + transpose: src (batch x K x N row-major fp32)
//   -> dst (batch x N x K bf16) so GEMM tiles are K-contiguous per column.
// ---------------------------------------------------------------------------
__global__ void cvt_t_k(const float* __restrict__ src, __bf16* __restrict__ dst,
                        int K, int Ncols, long long total)
{
    long long idx = (long long)blockIdx.x * blockDim.x + threadIdx.x;
    if (idx >= total) return;
    long long kn = (long long)K * Ncols;
    int b = (int)(idx / kn);
    int rem = (int)(idx - (long long)b * kn);
    int k = rem / Ncols;
    int n = rem - k * Ncols;
    dst[(long long)b * kn + (long long)n * K + k] = (__bf16)src[idx];
}

// ---------------------------------------------------------------------------
// Fused GEMM: C = epilogue( A(M x K) @ W(K x Ncols) + bias ) [+ resid]
//   A: fp32 or bf16 row-major; W: bf16 pre-transposed [Ncols][K]; C: fp32/bf16.
//   Block = 256 thr = 8 waves; wave -> 16 x 128 strip (8 accumulators).
//   W tiles are DMA'd into double-buffered LDS by the Tensor Data Mover
//   (wave 0 issues, TENSORcnt-tracked, overlapped with WMMA on the other
//   buffer); fallback is cooperative b128 staging. All fragment traffic is
//   128-bit. EXEC is all-ones at every v_wmma (tails via predicated ld/st).
// ---------------------------------------------------------------------------
template <bool ABF, bool OBF>
__global__ __launch_bounds__(256)
void gemm_k(const void* __restrict__ Ap, int lda,
            const __bf16* __restrict__ Wt,
            const float* __restrict__ bias,
            const float* __restrict__ resid,
            void* __restrict__ Cp, int ldc,
            int M, int K, int Ncols, int act)
{
    __shared__ __bf16 lds[2 * TILE_N * KSTR];

    const int tid  = threadIdx.x;
    const int wave = tid >> 5;
    const int lane = tid & 31;
    const int half = lane >> 4;
    const int l16  = lane & 15;

    const int rowBase = blockIdx.x * TILE_M + wave * 16;
    const int nBase   = blockIdx.y * TILE_N;
    const int row     = rowBase + l16;
    const bool rok    = (row < M);

    const float*  Af = (const float*)Ap;
    const __bf16* Ab = (const __bf16*)Ap;

    v8f acc[8] = {};

#if USE_TDM
    // LDS byte address of the staging buffers (flat address low bits = LDS offset;
    // this kernel's only static LDS allocation starts at offset 0).
    const unsigned ldsBase = (unsigned)(size_t)(void*)&lds[0];
    if (wave == 0)
        tdm_issue_w_tile(ldsBase, (unsigned long long)(Wt + (size_t)nBase * K),
                         K, Ncols);
#else
    // cooperative staging map: 256 threads -> 128 cols x 2 half-chunks of 16 bf16
    const int wc = tid >> 1;
    const int wh = (tid & 1) * 16;
#endif

    int it = 0;
    for (int k0 = 0; k0 < K; k0 += 32, ++it) {
        const int cur = it & 1;
        __bf16* tile = lds + cur * (TILE_N * KSTR);

#if USE_TDM
        if (wave == 0) __builtin_amdgcn_s_wait_tensorcnt((short)0);
        __syncthreads();   // publish tile `cur`; also proves prior reads of the
                           // other buffer are done before it is DMA'd again
        if (wave == 0 && (k0 + 32) < K)
            tdm_issue_w_tile(ldsBase + (unsigned)((1 - cur) * TILE_N * KSTR * 2),
                             (unsigned long long)(Wt + (size_t)nBase * K + k0 + 32),
                             K, Ncols);
#else
        __syncthreads();   // prior compute on this buffer (2 iters ago) is done
        {
            const __bf16* g = Wt + (size_t)(nBase + wc) * K + k0 + wh;
            *(v8bf*)(&tile[wc * KSTR + wh])     = *(const v8bf*)g;
            *(v8bf*)(&tile[wc * KSTR + wh + 8]) = *(const v8bf*)(g + 8);
        }
        __syncthreads();
#endif

        // A fragment (16x32): lane row = l16; K runs [half*8,+8) and [16+half*8,+8)
        v16bf a;
        if (ABF) {
            const __bf16* ap = Ab + (size_t)row * lda + k0 + half * 8;
            v8bf lo = {}, hi = {};
            if (rok) { lo = *(const v8bf*)ap; hi = *(const v8bf*)(ap + 16); }
            a = __builtin_shufflevector(lo, hi, 0,1,2,3,4,5,6,7,8,9,10,11,12,13,14,15);
        } else {
            const float* ap = Af + (size_t)row * lda + k0 + half * 8;
            float4 f0 = {0,0,0,0}, f1 = {0,0,0,0}, f2 = {0,0,0,0}, f3 = {0,0,0,0};
            if (rok) {
                f0 = *(const float4*)ap;        f1 = *(const float4*)(ap + 4);
                f2 = *(const float4*)(ap + 16); f3 = *(const float4*)(ap + 20);
            }
            a[0]=(__bf16)f0.x;  a[1]=(__bf16)f0.y;  a[2]=(__bf16)f0.z;  a[3]=(__bf16)f0.w;
            a[4]=(__bf16)f1.x;  a[5]=(__bf16)f1.y;  a[6]=(__bf16)f1.z;  a[7]=(__bf16)f1.w;
            a[8]=(__bf16)f2.x;  a[9]=(__bf16)f2.y;  a[10]=(__bf16)f2.z; a[11]=(__bf16)f2.w;
            a[12]=(__bf16)f3.x; a[13]=(__bf16)f3.y; a[14]=(__bf16)f3.z; a[15]=(__bf16)f3.w;
        }

        if (rok && (k0 + 32) < K) {
            const void* nxt = ABF ? (const void*)(Ab + (size_t)row * lda + k0 + 32)
                                  : (const void*)(Af + (size_t)row * lda + k0 + 32);
            __builtin_prefetch(nxt, 0, 1);
        }

        #pragma unroll
        for (int t = 0; t < 8; ++t) {
            int col = t * 16 + l16;   // lane holds B column
            v8bf blo = *(const v8bf*)(&tile[col * KSTR + half * 8]);
            v8bf bhi = *(const v8bf*)(&tile[col * KSTR + 16 + half * 8]);
            v16bf b = __builtin_shufflevector(blo, bhi,
                       0,1,2,3,4,5,6,7,8,9,10,11,12,13,14,15);
            acc[t] = __builtin_amdgcn_wmma_f32_16x16x32_bf16(
                false, a, false, b, (short)0, acc[t], false, false);
        }
    }

    // Epilogue: bias + activation (+ residual), fused store (fp32 or bf16)
    #pragma unroll
    for (int t = 0; t < 8; ++t) {
        int col = nBase + t * 16 + l16;
        float bv = bias ? bias[col] : 0.0f;
        #pragma unroll
        for (int r = 0; r < 8; ++r) {
            int rr = rowBase + r + 8 * half;
            if (rr < M) {
                float x = acc[t][r] + bv;
                if (act == ACT_SSP)       x = sspf(x);
                else if (act == ACT_RELU) x = fmaxf(x, 0.0f);
                if (resid) x += resid[(size_t)rr * ldc + col];
                if (OBF) ((__bf16*)Cp)[(size_t)rr * ldc + col] = (__bf16)x;
                else     ((float*)Cp)[(size_t)rr * ldc + col]  = x;
            }
        }
    }
}

// ---------------------------------------------------------------------------
// Node embedding (fp32, F=28 too small for WMMA)
// ---------------------------------------------------------------------------
__global__ void node_embed_k(const float* __restrict__ rf, const float* __restrict__ pf,
                             const int* __restrict__ atype,
                             const float* __restrict__ emb, const float* __restrict__ Wf,
                             float* __restrict__ z, int N)
{
    long long idx = (long long)blockIdx.x * blockDim.x + threadIdx.x;
    if (idx >= (long long)N * HDIM) return;
    int n = (int)(idx >> 8);
    int j = (int)(idx & 255);
    const float* r = rf + (long long)n * FDIM;
    if (j < 128) {
        float s = emb[(long long)atype[n] * 128 + j];
        #pragma unroll
        for (int f = 0; f < FDIM; ++f) s += r[f] * Wf[f * 128 + j];
        z[idx] = s;
    } else {
        const float* p = pf + (long long)n * FDIM;
        int jj = j - 128;
        float s = 0.0f;
        #pragma unroll
        for (int f = 0; f < FDIM; ++f) s += (p[f] - r[f]) * Wf[f * 128 + jj];
        z[idx] = s;
    }
}

__global__ void edge_geo_k(const float* __restrict__ pos,
                           const int* __restrict__ src, const int* __restrict__ dst,
                           float* __restrict__ elen, int E)
{
    int e = blockIdx.x * blockDim.x + threadIdx.x;
    if (e >= E) return;
    int s = src[e], d = dst[e];
    float dx = pos[3 * s + 0] - pos[3 * d + 0];
    float dy = pos[3 * s + 1] - pos[3 * d + 1];
    float dz = pos[3 * s + 2] - pos[3 * d + 2];
    elen[e] = sqrtf(dx * dx + dy * dy + dz * dz + 1e-12f);
}

// s[e][j] = ssp(elen[e]*W1[j] + b1[j])  -> bf16
__global__ void dist_mlp1_k(const float* __restrict__ elen,
                            const float* __restrict__ W1, const float* __restrict__ b1,
                            __bf16* __restrict__ out, int E)
{
    long long idx = (long long)blockIdx.x * blockDim.x + threadIdx.x;
    if (idx >= (long long)E * HDIM) return;
    int e = (int)(idx >> 8), j = (int)(idx & 255);
    out[idx] = (__bf16)sspf(elen[e] * W1[j] + b1[j]);
}

// cat[e] = [d_emb*bond[type_r], d_emb*bond[type_p]]  (bf16, E x 2H)
__global__ void attr_cat_k(const __bf16* __restrict__ demb, const float* __restrict__ bond,
                           const int* __restrict__ tr, const int* __restrict__ tp,
                           __bf16* __restrict__ cat, int E)
{
    long long idx = (long long)blockIdx.x * blockDim.x + threadIdx.x;
    if (idx >= (long long)E * HDIM) return;
    int e = (int)(idx >> 8), j = (int)(idx & 255);
    float v = (float)demb[idx];
    cat[(long long)e * 512 + j]       = (__bf16)(v * bond[(long long)tr[e] * HDIM + j]);
    cat[(long long)e * 512 + 256 + j] = (__bf16)(v * bond[(long long)tp[e] * HDIM + j]);
}

// cat[e] = [h[src]*h[dst], edge_attr]  (bf16, E x 2H)
__global__ void pair_cat_k(const float* __restrict__ h, const __bf16* __restrict__ ea,
                           const int* __restrict__ src, const int* __restrict__ dst,
                           __bf16* __restrict__ cat, int E)
{
    long long idx = (long long)blockIdx.x * blockDim.x + threadIdx.x;
    if (idx >= (long long)E * HDIM) return;
    int e = (int)(idx >> 8), j = (int)(idx & 255);
    cat[(long long)e * 512 + j] = (__bf16)(h[(long long)src[e] * HDIM + j] *
                                           h[(long long)dst[e] * HDIM + j]);
    cat[(long long)e * 512 + 256 + j] = ea[idx];
}

__global__ void fill0_k(float* __restrict__ p, long long n) {
    long long i = (long long)blockIdx.x * blockDim.x + threadIdx.x;
    if (i < n) p[i] = 0.0f;
}

// agg[dst] += hl1[src] * Wf[e]   (bf16 reads, native global_atomic_add_f32)
__global__ void scatter_msg_k(const __bf16* __restrict__ hl1, const __bf16* __restrict__ wf,
                              const int* __restrict__ src, const int* __restrict__ dst,
                              float* __restrict__ agg, int E)
{
    const int J = HDIM / 4;
    long long idx = (long long)blockIdx.x * blockDim.x + threadIdx.x;
    if (idx >= (long long)E * J) return;
    int e  = (int)(idx / J);
    int j4 = (int)(idx % J) * 4;
    int s = src[e], d = dst[e];
    const __bf16* ap = hl1 + (long long)s * HDIM + j4;
    const __bf16* wp = wf  + (long long)e * HDIM + j4;
    float* p = agg + (long long)d * HDIM + j4;
    #pragma unroll
    for (int i = 0; i < 4; ++i) {
        float v = (float)ap[i] * (float)wp[i];
        __hip_atomic_fetch_add(p + i, v, __ATOMIC_RELAXED, __HIP_MEMORY_SCOPE_AGENT);
    }
}

// final (E x 128 bf16) @ (128 x 1 fp32): one wave per edge, shuffle reduction
__global__ __launch_bounds__(256)
void final_dot_k(const __bf16* __restrict__ X, const float* __restrict__ w3,
                 const float* __restrict__ b3, float* __restrict__ out, int E)
{
    int wave = threadIdx.x >> 5, lane = threadIdx.x & 31;
    int e = blockIdx.x * 8 + wave;
    if (e >= E) return;
    float s = 0.0f;
    #pragma unroll
    for (int i = 0; i < 4; ++i) {
        int j = lane + i * 32;
        s += (float)X[(long long)e * 128 + j] * w3[j];
    }
    #pragma unroll
    for (int m = 16; m >= 1; m >>= 1) s += __shfl_xor(s, m, 32);
    if (lane == 0) out[e] = s + b3[0];
}

// ---------------------------------------------------------------------------
// Host orchestration
// ---------------------------------------------------------------------------
extern "C" void kernel_launch(void* const* d_in, const int* in_sizes, int n_in,
                              void* d_out, int out_size, void* d_ws, size_t ws_size,
                              hipStream_t stream)
{
    const int N = in_sizes[0];
    const int E = in_sizes[5];
    const int H = HDIM;
    const long long HH = (long long)H * H;
    const long long EH = (long long)E * H;
    const long long NH = (long long)N * H;

    const int*   atom_type = (const int*)  d_in[0];
    const float* r_feat    = (const float*)d_in[1];
    const float* p_feat    = (const float*)d_in[2];
    const float* pos       = (const float*)d_in[3];
    const int*   eidx      = (const int*)  d_in[4];
    const int*   etr       = (const int*)  d_in[5];
    const int*   etp       = (const int*)  d_in[6];
    const float* atom_emb  = (const float*)d_in[7];
    const float* feat_W    = (const float*)d_in[8];
    const float* bond_emb  = (const float*)d_in[9];
    const float* em_W1     = (const float*)d_in[10];
    const float* em_b1     = (const float*)d_in[11];
    const float* em_W2     = (const float*)d_in[12];
    const float* em_b2     = (const float*)d_in[13];
    const float* cat_W1    = (const float*)d_in[14];
    const float* cat_b1    = (const float*)d_in[15];
    const float* cat_W2    = (const float*)d_in[16];
    const float* cat_b2    = (const float*)d_in[17];
    const float* filt_W1   = (const float*)d_in[18];
    const float* filt_b1   = (const float*)d_in[19];
    const float* filt_W2   = (const float*)d_in[20];
    const float* filt_b2   = (const float*)d_in[21];
    const float* lin1_W    = (const float*)d_in[22];
    const float* lin2_W    = (const float*)d_in[23];
    const float* lin2_b    = (const float*)d_in[24];
    const float* lin3_W    = (const float*)d_in[25];
    const float* lin3_b    = (const float*)d_in[26];
    const float* grad_W1   = (const float*)d_in[27];
    const float* grad_b1   = (const float*)d_in[28];
    const float* grad_W2   = (const float*)d_in[29];
    const float* grad_b2   = (const float*)d_in[30];
    const float* grad_W3   = (const float*)d_in[31];
    const float* grad_b3   = (const float*)d_in[32];

    const int* src = eidx;
    const int* dst = eidx + E;
    float* out_inv = (float*)d_out;
    float* out_len = out_inv + E;

    // ---- workspace carve-up (each buffer 256B aligned) ----------------------
    char* wsp = (char*)d_ws;
    size_t off = 0;
    auto balloc = [&](long long nelem) {  // bf16
        __bf16* p = (__bf16*)(wsp + off);
        off = ((off + (size_t)nelem * 2 + 255) / 256) * 256;
        return p;
    };
    auto falloc = [&](long long nelem) {  // fp32
        float* p = (float*)(wsp + off);
        off = ((off + (size_t)nelem * 4 + 255) / 256) * 256;
        return p;
    };
    // transposed bf16 weights (~5 MB, L2-resident across all GEMMs)
    __bf16* bWm2  = balloc(HH);
    __bf16* bCat1 = balloc(2 * HH);
    __bf16* bCat2 = balloc(HH);
    __bf16* bF1   = balloc(LAYERS * HH);
    __bf16* bF2   = balloc(LAYERS * HH);
    __bf16* bL1   = balloc(LAYERS * HH);
    __bf16* bL2   = balloc(LAYERS * HH);
    __bf16* bL3   = balloc(LAYERS * HH);
    __bf16* bG1   = balloc(2 * HH);
    __bf16* bG2   = balloc((long long)H * 128);
    // bf16 activations
    __bf16* sA    = balloc(EH);        // GEMM chain scratch (E x H)
    __bf16* sCat  = balloc(2 * EH);    // concat inputs (E x 2H)
    __bf16* eAtrB = balloc(EH);        // edge_attr (persists across layers)
    __bf16* wfB   = balloc(EH);        // Wf / d_emb / grad-head (E x 128) reuse
    __bf16* hl1B  = balloc(NH);        // h @ lin1
    __bf16* ntB   = balloc(NH);        // ssp(agg@l2+b)
    // fp32 node tensors
    float* hA  = falloc(NH);
    float* hB  = falloc(NH);
    float* agg = falloc(NH);

    // ---- helpers ------------------------------------------------------------
    auto cvt = [&](const float* s, __bf16* d, int K, int Ncols, int batch) {
        long long total = (long long)batch * K * Ncols;
        cvt_t_k<<<(int)((total + 255) / 256), 256, 0, stream>>>(s, d, K, Ncols, total);
    };
    auto ew_blocks = [&](long long n) { return (int)((n + 255) / 256); };

    #define GEMM(ABF, OBF, A, lda, Wt, Ncols, bias, resid, C, ldc, M, K, act)        \
        gemm_k<ABF, OBF><<<dim3(((M) + TILE_M - 1) / TILE_M, (Ncols) / TILE_N),      \
                           256, 0, stream>>>((A), (lda), (Wt), (bias), (resid),      \
                                             (C), (ldc), (M), (K), (Ncols), (act))

    // ---- 0. convert + transpose weights -------------------------------------
    cvt(em_W2,   bWm2,  H, H, 1);
    cvt(cat_W1,  bCat1, 2 * H, H, 1);
    cvt(cat_W2,  bCat2, H, H, 1);
    cvt(filt_W1, bF1,   H, H, LAYERS);
    cvt(filt_W2, bF2,   H, H, LAYERS);
    cvt(lin1_W,  bL1,   H, H, LAYERS);
    cvt(lin2_W,  bL2,   H, H, LAYERS);
    cvt(lin3_W,  bL3,   H, H, LAYERS);
    cvt(grad_W1, bG1,   2 * H, H, 1);
    cvt(grad_W2, bG2,   H, 128, 1);

    // ---- 1. node embedding --------------------------------------------------
    node_embed_k<<<ew_blocks(NH), 256, 0, stream>>>(r_feat, p_feat, atom_type,
                                                    atom_emb, feat_W, hA, N);

    // ---- 2. edge lengths + edge encoder -------------------------------------
    edge_geo_k<<<(E + 255) / 256, 256, 0, stream>>>(pos, src, dst, out_len, E);
    dist_mlp1_k<<<ew_blocks(EH), 256, 0, stream>>>(out_len, em_W1, em_b1, sA, E);
    GEMM(true, true, sA, H, bWm2, H, em_b2, nullptr, wfB, H, E, H, ACT_NONE);  // d_emb
    attr_cat_k<<<ew_blocks(EH), 256, 0, stream>>>(wfB, bond_emb, etr, etp, sCat, E);
    GEMM(true, true, sCat, 2 * H, bCat1, H, cat_b1, nullptr, sA, H, E, 2 * H, ACT_RELU);
    GEMM(true, true, sA, H, bCat2, H, cat_b2, nullptr, eAtrB, H, E, H, ACT_NONE); // edge_attr

    // ---- 3. SchNet interaction layers ---------------------------------------
    float* hcur = hA;
    float* hnxt = hB;
    for (int l = 0; l < LAYERS; ++l) {
        const __bf16* fW1 = bF1 + (long long)l * HH;
        const __bf16* fW2 = bF2 + (long long)l * HH;
        const __bf16* w1  = bL1 + (long long)l * HH;
        const __bf16* w2  = bL2 + (long long)l * HH;
        const __bf16* w3  = bL3 + (long long)l * HH;
        const float* fb1 = filt_b1 + (long long)l * H;
        const float* fb2 = filt_b2 + (long long)l * H;
        const float* l2b = lin2_b  + (long long)l * H;
        const float* l3b = lin3_b  + (long long)l * H;

        GEMM(true, true, eAtrB, H, fW1, H, fb1, nullptr, sA,  H, E, H, ACT_SSP);
        GEMM(true, true, sA,    H, fW2, H, fb2, nullptr, wfB, H, E, H, ACT_NONE); // Wf
        GEMM(false, true, hcur, H, w1,  H, nullptr, nullptr, hl1B, H, N, H, ACT_NONE);
        fill0_k<<<ew_blocks(NH), 256, 0, stream>>>(agg, NH);
        scatter_msg_k<<<ew_blocks((long long)E * (H / 4)), 256, 0, stream>>>(
            hl1B, wfB, src, dst, agg, E);
        GEMM(false, true, agg, H, w2, H, l2b, nullptr, ntB, H, N, H, ACT_SSP);
        GEMM(true, false, ntB, H, w3, H, l3b, hcur, hnxt, H, N, H, ACT_NONE);  // +resid
        float* t = hcur; hcur = hnxt; hnxt = t;
    }

    // ---- 4. grad head -------------------------------------------------------
    pair_cat_k<<<ew_blocks(EH), 256, 0, stream>>>(hcur, eAtrB, src, dst, sCat, E);
    GEMM(true, true, sCat, 2 * H, bG1, H, grad_b1, nullptr, sA, H, E, 2 * H, ACT_RELU);
    GEMM(true, true, sA, H, bG2, 128, grad_b2, nullptr, wfB, 128, E, H, ACT_RELU);
    final_dot_k<<<(E + 7) / 8, 256, 0, stream>>>(wfB, grad_W3, grad_b3, out_inv, E);

    #undef GEMM
}
